// NSGP_53128745452107
// MI455X (gfx1250) — compile-verified, matrix operand order checked by reading, not run on hardware
//
#include <hip/hip_runtime.h>
#include <hip/hip_bf16.h>
#include <math.h>

// ---------------------------------------------------------------------------
// NSGP marginal-likelihood for MI455X (gfx1250, wave32, WMMA).
// N=2048, D=5, M=64. Only the LAST dim's posterior Cholesky contributes (B),
// so we do exactly two 2048x2048 Choleskys (posterior dim-4 + global K).
// Rank-64 SYRK/GEMM work runs on v_wmma_f32_16x16x32_f16 with 2x2 register
// blocking (32x32 tile / wave, 8 WMMAs, every fragment reused twice).
// ---------------------------------------------------------------------------

#define NN   2048
#define DD   5
#define MM   64
#define NB   64
#define JIT  1e-3f
#define PI_F 3.14159265358979323846f

typedef __attribute__((ext_vector_type(16))) _Float16 v16h;
typedef __attribute__((ext_vector_type(8)))  float    v8f;

// --- fragment loader: row-major P (ld=64, f16). Per CDNA5 ISA 16-bit A 16x32
// layout: lane L holds row M=L&15; lanes<16 get K {0..7,16..23}, lanes>=16 get
// K {8..15,24..31}. Halves 0..7 -> K base+0..7, halves 8..15 -> K base+16..23.
// B (32x16) mirrors with lane = N, so the same loader serves both operands of
// the symmetric product P·P^T. Lowers to two global_load_b128 per fragment.
__device__ inline v16h load_frag64(const _Float16* __restrict__ P, int rowBase, int chunk) {
    int lane = threadIdx.x & 31;
    const _Float16* p = P + (size_t)(rowBase + (lane & 15)) * 64
                          + chunk * 32 + ((lane & 16) ? 8 : 0);
    v16h f;
#pragma unroll
    for (int t = 0; t < 8; ++t) { f[t] = p[t]; f[8 + t] = p[16 + t]; }
    return f;
}

// 32x32 tile of P(i0:,:64) * P(j0:,:64)^T with 2x2 register blocking.
// c[0]=(i0,j0) c[1]=(i0,j0+16) c[2]=(i0+16,j0) c[3]=(i0+16,j0+16)
__device__ inline void ppT_tile32(const _Float16* __restrict__ P, int i0, int j0, v8f c[4]) {
#pragma unroll
    for (int chunk = 0; chunk < 2; ++chunk) {
        v16h a0 = load_frag64(P, i0,      chunk);
        v16h a1 = load_frag64(P, i0 + 16, chunk);
        v16h b0 = load_frag64(P, j0,      chunk);
        v16h b1 = load_frag64(P, j0 + 16, chunk);
        c[0] = __builtin_amdgcn_wmma_f32_16x16x32_f16(false, a0, false, b0, (short)0, c[0], false, false);
        c[1] = __builtin_amdgcn_wmma_f32_16x16x32_f16(false, a0, false, b1, (short)0, c[1], false, false);
        c[2] = __builtin_amdgcn_wmma_f32_16x16x32_f16(false, a1, false, b0, (short)0, c[2], false, false);
        c[3] = __builtin_amdgcn_wmma_f32_16x16x32_f16(false, a1, false, b1, (short)0, c[3], false, false);
    }
}

// ---------------------------------------------------------------------------
// 1) Per-dim 64x64 kernel matrix, Cholesky, alpha = K^{-1} log(lls).
//    One 64-thread workgroup per dim; whole factorization in LDS.
// ---------------------------------------------------------------------------
__global__ void dim_setup(const float* __restrict__ Xbar, const float* __restrict__ lls,
                          const float* __restrict__ gls,  const float* __restrict__ gstd,
                          const float* __restrict__ gnoise,
                          float* __restrict__ Ld, float* __restrict__ alphad) {
    __shared__ float T[64][65];
    __shared__ float z[64];
    __shared__ float al[64];
    int d = blockIdx.x, tid = threadIdx.x;
    float ls = gls[d], st = gstd[d], nz = gnoise[d];
    float inv2 = -0.5f / (ls * ls), s2 = st * st;
    float xi = Xbar[d * MM + tid];
    for (int j = 0; j < 64; ++j) {
        float dx = xi - Xbar[d * MM + j];
        float v = s2 * expf(inv2 * dx * dx);
        if (j == tid) v += nz * nz + JIT;
        T[tid][j] = v;
    }
    __syncthreads();
    for (int kk = 0; kk < 64; ++kk) {                 // right-looking Cholesky
        if (tid == kk) T[kk][kk] = sqrtf(T[kk][kk]);
        __syncthreads();
        if (tid > kk) T[tid][kk] /= T[kk][kk];
        __syncthreads();
        if (tid > kk) {
            float lik = T[tid][kk];
            for (int j = kk + 1; j <= tid; ++j) T[tid][j] -= lik * T[j][kk];
        }
        __syncthreads();
    }
    if (tid == 0) {                                   // L z = log(lls); L^T a = z
        for (int kk = 0; kk < 64; ++kk) {
            float s = logf(lls[d * MM + kk]);
            for (int j = 0; j < kk; ++j) s -= T[kk][j] * z[j];
            z[kk] = s / T[kk][kk];
        }
        for (int kk = 63; kk >= 0; --kk) {
            float s = z[kk];
            for (int j = kk + 1; j < 64; ++j) s -= T[j][kk] * al[j];
            al[kk] = s / T[kk][kk];
        }
    }
    __syncthreads();
    for (int j = 0; j < 64; ++j) Ld[(size_t)d * 4096 + tid * 64 + j] = T[tid][j];
    alphad[d * MM + tid] = al[tid];
}

// ---------------------------------------------------------------------------
// 2) l_all[d][i] = exp(k_star(i,:) . alpha_d); stash k_star rows for last dim.
// ---------------------------------------------------------------------------
__global__ void kstar_l(const float* __restrict__ X, const float* __restrict__ Xbar,
                        const float* __restrict__ alphad,
                        const float* __restrict__ gls, const float* __restrict__ gstd,
                        float* __restrict__ l_all, float* __restrict__ kstar_last, int n) {
    int i = blockIdx.x * blockDim.x + threadIdx.x;
    int d = blockIdx.y;
    if (i >= n) return;
    float xi = X[(size_t)i * DD + d];
    float ls = gls[d], st = gstd[d];
    float inv2 = -0.5f / (ls * ls), s2 = st * st;
    float acc = 0.f;
    bool last = (d == DD - 1);
    for (int m = 0; m < MM; ++m) {
        float dx = xi - Xbar[d * MM + m];
        float kv = s2 * expf(inv2 * dx * dx);
        acc += kv * alphad[d * MM + m];
        if (last) kstar_last[(size_t)i * MM + m] = kv;
    }
    l_all[(size_t)d * n + i] = expf(acc);
}

// ---------------------------------------------------------------------------
// 3) W = k_star L^{-T}: per-row forward solve against L4; emit f16 for WMMA.
// ---------------------------------------------------------------------------
__global__ void wsolve(const float* __restrict__ Ld, const float* __restrict__ kstar_last,
                       _Float16* __restrict__ Ph, int n) {
    __shared__ float Ls[64][65];
    int tid = threadIdx.x;
    for (int j = 0; j < 64; ++j) Ls[tid][j] = Ld[(size_t)(DD - 1) * 4096 + tid * 64 + j];
    __syncthreads();
    int row = blockIdx.x * 64 + tid;
    float a[64];
    for (int c = 0; c < 64; ++c) a[c] = kstar_last[(size_t)row * MM + c];
    for (int c = 0; c < 64; ++c) {
        float s = a[c];
        for (int t = 0; t < c; ++t) s -= Ls[c][t] * a[t];
        a[c] = s / Ls[c][c];
    }
    for (int c = 0; c < 64; ++c) Ph[(size_t)row * 64 + c] = (_Float16)a[c];
}

// ---------------------------------------------------------------------------
// 4) k_post = k_ss - W W^T + 10*jitter*I  (32x32 tile / wave, 8 WMMAs)
// ---------------------------------------------------------------------------
__global__ void build_kpost(const float* __restrict__ X, const _Float16* __restrict__ Ph,
                            const float* __restrict__ gls, const float* __restrict__ gstd,
                            float* __restrict__ A, int n) {
    int tiles = n / 32;
    int i0 = (blockIdx.x / tiles) * 32, j0 = (blockIdx.x % tiles) * 32;
    v8f c[4] = {};
    ppT_tile32(Ph, i0, j0, c);
    int lane = threadIdx.x & 31;
    int Nn = lane & 15, Mb = (lane & 16) ? 8 : 0;
    float ls = gls[DD - 1], st = gstd[DD - 1];
    float inv2 = -0.5f / (ls * ls), s2 = st * st;
#pragma unroll
    for (int q = 0; q < 4; ++q) {
        int ib = i0 + ((q >> 1) ? 16 : 0);
        int jb = j0 + ((q & 1) ? 16 : 0);
#pragma unroll
        for (int v = 0; v < 8; ++v) {
            int i = ib + Mb + v, j = jb + Nn;
            float dx = X[(size_t)i * DD + (DD - 1)] - X[(size_t)j * DD + (DD - 1)];
            float val = s2 * expf(inv2 * dx * dx) - c[q][v];
            if (i == j) val += 10.f * JIT;
            A[(size_t)i * n + j] = val;
        }
    }
}

// ---------------------------------------------------------------------------
// 5) global K: prefix * exp(-ssd) with periodic term on the time dim.
// ---------------------------------------------------------------------------
__global__ void build_kglobal(const float* __restrict__ X, const float* __restrict__ l_all,
                              const float* __restrict__ g_std, const float* __restrict__ g_noise,
                              const float* __restrict__ period, float* __restrict__ K, int n) {
    size_t idx = (size_t)blockIdx.x * blockDim.x + threadIdx.x;
    int i = (int)(idx / n), j = (int)(idx % n);
    float prefix = 1.f, ssd = 0.f;
    float invp = PI_F / period[0];
#pragma unroll
    for (int d = 0; d < DD; ++d) {
        float diff = X[(size_t)i * DD + d] - X[(size_t)j * DD + d];
        float li = l_all[(size_t)d * n + i], lj = l_all[(size_t)d * n + j];
        float lsq = li * li + lj * lj;
        prefix *= sqrtf(2.f * li * lj / lsq);
        float num = diff * diff;
        if (d == DD - 1) {
            float s = sinf(invp * fabsf(diff));
            num += 4.f * s * s;
        }
        ssd += num / lsq;
    }
    float gs = g_std[0], gn = g_noise[0];
    float val = gs * gs * prefix * expf(-ssd);
    if (i == j) val += gn * gn + JIT;
    K[idx] = val;
}

// ---------------------------------------------------------------------------
// 6) blocked Cholesky: potrf(64x64 diag in LDS) -> trsm(panel rows) ->
//    WMMA SYRK trailing update (32x32 tiles, lower triangle only).
// ---------------------------------------------------------------------------
__global__ void potrf_diag(float* __restrict__ A, int n, int k) {
    __shared__ float T[64][65];
    int tid = threadIdx.x;
    size_t base = (size_t)(k * NB) * n + k * NB;
    for (int j = 0; j < 64; ++j) T[tid][j] = A[base + (size_t)tid * n + j];
    __syncthreads();
    for (int kk = 0; kk < 64; ++kk) {
        if (tid == kk) T[kk][kk] = sqrtf(T[kk][kk]);
        __syncthreads();
        if (tid > kk) T[tid][kk] /= T[kk][kk];
        __syncthreads();
        if (tid > kk) {
            float lik = T[tid][kk];
            for (int j = kk + 1; j <= tid; ++j) T[tid][j] -= lik * T[j][kk];
        }
        __syncthreads();
    }
    for (int j = 0; j <= tid; ++j) A[base + (size_t)tid * n + j] = T[tid][j];
}

__global__ void trsm_panel(float* __restrict__ A, _Float16* __restrict__ Ph, int n, int k) {
    __shared__ float Ls[64][65];
    int tid = threadIdx.x;
    size_t dbase = (size_t)(k * NB) * n + k * NB;
    for (int j = 0; j < 64; ++j) Ls[tid][j] = A[dbase + (size_t)tid * n + j];
    __syncthreads();
    int row = (k + 1) * NB + blockIdx.x * 64 + tid;
    size_t rbase = (size_t)row * n + k * NB;
    float a[64];
    for (int c = 0; c < 64; ++c) a[c] = A[rbase + c];
    for (int c = 0; c < 64; ++c) {
        float s = a[c];
        for (int t = 0; t < c; ++t) s -= Ls[c][t] * a[t];
        a[c] = s / Ls[c][c];
    }
    for (int c = 0; c < 64; ++c) {
        A[rbase + c] = a[c];
        Ph[(size_t)row * 64 + c] = (_Float16)a[c];
    }
}

__global__ void syrk_update(float* __restrict__ A, const _Float16* __restrict__ Ph, int n, int k) {
    int trail = (k + 1) * NB;
    int tiles = (n - trail) / 32;
    int i0 = trail + (blockIdx.x / tiles) * 32;
    int j0 = trail + (blockIdx.x % tiles) * 32;
    if (i0 < j0) return;                 // wave-uniform: EXEC stays all-ones for WMMA
    v8f c[4] = {};
    ppT_tile32(Ph, i0, j0, c);
    int lane = threadIdx.x & 31;
    int Nn = lane & 15, Mb = (lane & 16) ? 8 : 0;
#pragma unroll
    for (int q = 0; q < 4; ++q) {
        int ib = i0 + ((q >> 1) ? 16 : 0);
        int jb = j0 + ((q & 1) ? 16 : 0);
#pragma unroll
        for (int v = 0; v < 8; ++v) {
            size_t p = (size_t)(ib + Mb + v) * n + (jb + Nn);
            A[p] -= c[q][v];
        }
    }
}

// ---------------------------------------------------------------------------
// 7) reductions + triangular solves (single-workgroup; trivial FLOP count)
// ---------------------------------------------------------------------------
__global__ void sum_log_diag(const float* __restrict__ Mt, int n, float* __restrict__ out) {
    __shared__ float red[256];
    int tid = threadIdx.x;
    float s = 0.f;
    for (int i = tid; i < n; i += 256) s += logf(Mt[(size_t)i * n + i]);
    red[tid] = s; __syncthreads();
    for (int off = 128; off > 0; off >>= 1) { if (tid < off) red[tid] += red[tid + off]; __syncthreads(); }
    if (tid == 0) out[0] = red[0];
}

__global__ void trisolve(const float* __restrict__ L, const float* __restrict__ b,
                         float* __restrict__ x, int n, int trans) {
    __shared__ float xs[NN];
    __shared__ float red[256];
    int tid = threadIdx.x;
    if (!trans) {
        for (int i = 0; i < n; ++i) {
            float s = 0.f;
            for (int j = tid; j < i; j += 256) s += L[(size_t)i * n + j] * xs[j];
            red[tid] = s; __syncthreads();
            for (int off = 128; off > 0; off >>= 1) { if (tid < off) red[tid] += red[tid + off]; __syncthreads(); }
            if (tid == 0) xs[i] = (b[i] - red[0]) / L[(size_t)i * n + i];
            __syncthreads();
        }
    } else {
        for (int i = n - 1; i >= 0; --i) {
            float s = 0.f;
            for (int j = i + 1 + tid; j < n; j += 256) s += L[(size_t)j * n + i] * xs[j];
            red[tid] = s; __syncthreads();
            for (int off = 128; off > 0; off >>= 1) { if (tid < off) red[tid] += red[tid + off]; __syncthreads(); }
            if (tid == 0) xs[i] = (b[i] - red[0]) / L[(size_t)i * n + i];
            __syncthreads();
        }
    }
    for (int i = tid; i < n; i += 256) x[i] = xs[i];
}

__global__ void final_combine(const float* __restrict__ y, const float* __restrict__ alpha,
                              const float* __restrict__ logdetG, const float* __restrict__ logdetPost,
                              float* __restrict__ out, int n) {
    __shared__ float red[256];
    int tid = threadIdx.x;
    float s = 0.f;
    for (int i = tid; i < n; i += 256) s += y[i] * alpha[i];
    red[tid] = s; __syncthreads();
    for (int off = 128; off > 0; off >>= 1) { if (tid < off) red[tid] += red[tid + off]; __syncthreads(); }
    if (tid == 0) {
        float A = 0.5f * (red[0] + logdetG[0]);
        out[0] = (A + logdetPost[0]) / (float)(n * DD);
    }
}

// ---------------------------------------------------------------------------
// host driver: deterministic launch sequence, all on `stream`, scratch in d_ws
// ---------------------------------------------------------------------------
extern "C" void kernel_launch(void* const* d_in, const int* in_sizes, int n_in,
                              void* d_out, int out_size, void* d_ws, size_t ws_size,
                              hipStream_t stream) {
    const int n = NN;
    const float* X      = (const float*)d_in[0];
    const float* yv     = (const float*)d_in[1];
    const float* Xbar   = (const float*)d_in[2];
    const float* lls    = (const float*)d_in[3];
    const float* gls    = (const float*)d_in[4];
    const float* gstd   = (const float*)d_in[5];
    const float* gnoise = (const float*)d_in[6];
    const float* g_std  = (const float*)d_in[7];
    const float* g_noise= (const float*)d_in[8];
    const float* period = (const float*)d_in[9];

    char* ws = (char*)d_ws;
    auto alloc = [&](size_t bytes) -> char* {
        char* p = ws;
        ws += (bytes + 255) & ~(size_t)255;
        return p;
    };
    float*    l_all   = (float*)   alloc((size_t)DD * n * 4);
    float*    kstar4  = (float*)   alloc((size_t)n * MM * 4);
    float*    Ld      = (float*)   alloc((size_t)DD * 4096 * 4);
    float*    alphad  = (float*)   alloc((size_t)DD * MM * 4);
    _Float16* Ph      = (_Float16*)alloc((size_t)n * 64 * 2);
    float*    Amat    = (float*)   alloc((size_t)n * n * 4);   // posterior (dim 4)
    float*    KG      = (float*)   alloc((size_t)n * n * 4);   // global K
    float*    zbuf    = (float*)   alloc((size_t)n * 4);
    float*    abuf    = (float*)   alloc((size_t)n * 4);
    float*    sc      = (float*)   alloc(4 * 4);               // [0]=logdetPost [1]=logdetG

    // per-dim small GP solves
    dim_setup<<<DD, 64, 0, stream>>>(Xbar, lls, gls, gstd, gnoise, Ld, alphad);
    dim3 g1(n / 256, DD);
    kstar_l<<<g1, 256, 0, stream>>>(X, Xbar, alphad, gls, gstd, l_all, kstar4, n);
    wsolve<<<n / 64, 64, 0, stream>>>(Ld, kstar4, Ph, n);

    // posterior covariance for the last dim, then its Cholesky
    build_kpost<<<(n / 32) * (n / 32), 32, 0, stream>>>(X, Ph, gls, gstd, Amat, n);
    for (int k = 0; k < n / NB; ++k) {
        potrf_diag<<<1, 64, 0, stream>>>(Amat, n, k);
        int rem = n - (k + 1) * NB;
        if (rem > 0) {
            trsm_panel<<<rem / 64, 64, 0, stream>>>(Amat, Ph, n, k);
            int t32 = rem / 32;
            syrk_update<<<t32 * t32, 32, 0, stream>>>(Amat, Ph, n, k);
        }
    }
    sum_log_diag<<<1, 256, 0, stream>>>(Amat, n, sc + 0);

    // global kernel matrix + Cholesky + solves
    build_kglobal<<<(int)(((size_t)n * n) / 256), 256, 0, stream>>>(X, l_all, g_std, g_noise, period, KG, n);
    for (int k = 0; k < n / NB; ++k) {
        potrf_diag<<<1, 64, 0, stream>>>(KG, n, k);
        int rem = n - (k + 1) * NB;
        if (rem > 0) {
            trsm_panel<<<rem / 64, 64, 0, stream>>>(KG, Ph, n, k);
            int t32 = rem / 32;
            syrk_update<<<t32 * t32, 32, 0, stream>>>(KG, Ph, n, k);
        }
    }
    sum_log_diag<<<1, 256, 0, stream>>>(KG, n, sc + 1);
    trisolve<<<1, 256, 0, stream>>>(KG, yv, zbuf, n, 0);
    trisolve<<<1, 256, 0, stream>>>(KG, zbuf, abuf, n, 1);
    final_combine<<<1, 256, 0, stream>>>(yv, abuf, sc + 1, sc + 0, (float*)d_out, n);
}
